// PinSageConv_88441966559451
// MI455X (gfx1250) — matrix operand order repaired and checked by reading, not compiled, not run on hardware
//
#include <hip/hip_runtime.h>
#include <math.h>

typedef __bf16 bf16;
typedef __attribute__((ext_vector_type(16))) __bf16 v16bf;
typedef __attribute__((ext_vector_type(8)))  __bf16 v8bf;
typedef __attribute__((ext_vector_type(8)))  float   v8f;
typedef __attribute__((ext_vector_type(4)))  float   f32x4;

static constexpr int DIN  = 128;   // input feature dim
static constexpr int DHID = 128;   // hidden dim (GEMM1 output)
static constexpr int DOUT = 128;   // output dim
static constexpr int DCAT = 256;   // concat dim
static constexpr int KNB  = 32;    // neighbors per node
static constexpr int NODES_PER_BLOCK = 16;
static constexpr int NODES_PER_WAVE  = 4;
static constexpr float NEG_SLOPE = 0.01f;

__device__ __forceinline__ float leaky(float v) {
    return v > 0.f ? v : NEG_SLOPE * v;
}

__global__ __launch_bounds__(128) void pinsage_fused_kernel(
    const float* __restrict__ h_node,   // [N,128]
    const float* __restrict__ h_ngbrs,  // [N,32,128]
    const float* __restrict__ alpha,    // [N,32]
    const float* __restrict__ Wq,       // [128,128]
    const float* __restrict__ bq,       // [128]
    const float* __restrict__ Ww,       // [128,256]
    const float* __restrict__ bw,       // [128]
    float* __restrict__ out,            // [N,128]
    int Nn)
{
    __shared__ __align__(64) bf16  Wq_lds[DHID * DIN];              // 32 KB, row-major [hid][in]
    __shared__ __align__(64) bf16  cat_lds[NODES_PER_BLOCK][DCAT];  // 8 KB  [node][concat]
    __shared__ float norm2[NODES_PER_BLOCK];

    const int tid  = threadIdx.x;
    const int lane = tid & 31;
    const int wave = tid >> 5;
    const int l16  = lane & 15;
    const int half = lane >> 4;
    const int blockBase = blockIdx.x * NODES_PER_BLOCK;

    // ---- Stage 0: Wq f32->bf16 into LDS; h_node into concat cols [0,128) ----
    for (int i = tid; i < DHID * DIN; i += 128)
        Wq_lds[i] = (bf16)Wq[i];
    for (int i = tid; i < NODES_PER_BLOCK * DIN; i += 128) {
        int ln = i >> 7, c = i & 127;
        int g = blockBase + ln;
        cat_lds[ln][c] = (g < Nn) ? (bf16)h_node[(size_t)g * DIN + c] : (bf16)0.f;
    }
    if (tid < NODES_PER_BLOCK) norm2[tid] = 0.f;
    __syncthreads();

    // bq fragment: column of N-tile nt for this lane is nt*16 + l16
    float bqv[8];
    #pragma unroll
    for (int nt = 0; nt < 8; ++nt) bqv[nt] = bq[nt * 16 + l16];

    // ---- Stage 1: per-wave neighbor GEMM + LeakyReLU + alpha-weighted mean ----
    for (int i = 0; i < NODES_PER_WAVE; ++i) {
        const int ln   = wave * NODES_PER_WAVE + i;  // 0..15
        const int node = blockBase + ln;
        if (node >= Nn) continue;                    // wave-uniform branch: EXEC stays full

        // sum of alphas (safediv denominator)
        float asum = alpha[(size_t)node * KNB + lane];
        #pragma unroll
        for (int m = 16; m >= 1; m >>= 1) asum += __shfl_xor(asum, m, 32);
        const float adenom = (asum == 0.f) ? 1.f : asum;

        float colsum[8];
        #pragma unroll
        for (int nt = 0; nt < 8; ++nt) colsum[nt] = 0.f;

        #pragma unroll
        for (int mt = 0; mt < 2; ++mt) {             // 2 M-tiles of 16 neighbor rows
            v8f acc[8];
            #pragma unroll
            for (int nt = 0; nt < 8; ++nt) acc[nt] = v8f{};

            #pragma unroll
            for (int ks = 0; ks < 4; ++ks) {         // K = 128 in steps of 32
                // A fragment (contiguous k order, same as B): lane row = mt*16 + l16,
                // slots s -> K = half*16 + s
                const float* arow =
                    h_ngbrs + ((size_t)node * KNB + mt * 16 + l16) * DIN + ks * 32;
                const f32x4* c0 = (const f32x4*)(arow + half * 16);
                const f32x4* c1 = (const f32x4*)(arow + half * 16 + 8);
                f32x4 p0 = c0[0], p1 = c0[1], p2 = c1[0], p3 = c1[1];
                v16bf afrag;
                #pragma unroll
                for (int e = 0; e < 4; ++e) {
                    afrag[e]      = (bf16)p0[e];
                    afrag[4 + e]  = (bf16)p1[e];
                    afrag[8 + e]  = (bf16)p2[e];
                    afrag[12 + e] = (bf16)p3[e];
                }
                #pragma unroll
                for (int nt = 0; nt < 8; ++nt) {
                    // B fragment: column n = nt*16 + l16 of Wq^T  ==  row of Wq;
                    // slots s -> K = half*16 + s  => 16 contiguous bf16 (2x ds_load_b128)
                    const v16bf bfrag = *(const v16bf*)
                        &Wq_lds[(nt * 16 + l16) * DIN + ks * 32 + half * 16];
                    acc[nt] = __builtin_amdgcn_wmma_f32_16x16x32_bf16(
                        false, afrag, false, bfrag, (short)0, acc[nt], false, false);
                }
            }
            // epilogue: bias + LeakyReLU, alpha-weighted column accumulation.
            // C/D layout: VGPR j holds row M = j + 8*half, column N = l16.
            #pragma unroll
            for (int j = 0; j < 8; ++j) {
                const int r = mt * 16 + half * 8 + j;            // neighbor index
                const float aw = alpha[(size_t)node * KNB + r];  // broadcast across l16
                #pragma unroll
                for (int nt = 0; nt < 8; ++nt) {
                    float v = leaky(acc[nt][j] + bqv[nt]);
                    colsum[nt] += aw * v;
                }
            }
        }
        // rows split across lane halves: fold, then safediv and stash h_agg (bf16)
        #pragma unroll
        for (int nt = 0; nt < 8; ++nt) {
            float t = colsum[nt] + __shfl_xor(colsum[nt], 16, 32);
            if (half == 0)
                cat_lds[ln][DIN + nt * 16 + l16] = (bf16)(t / adenom);
        }
    }
    __syncthreads();

    // ---- Stage 2: GEMM2  [16 x 256] x Ww^T -> [16 x 128]; wave owns N-tiles {2w,2w+1} ----
    const int nt0 = wave * 2;
    v8f acc2[2];
    acc2[0] = v8f{}; acc2[1] = v8f{};
    #pragma unroll
    for (int ks = 0; ks < 8; ++ks) {                 // K = 256 in steps of 32
        const bf16* arow = &cat_lds[l16][ks * 32];
        v8bf a_lo = *(const v8bf*)(arow + half * 16);
        v8bf a_hi = *(const v8bf*)(arow + half * 16 + 8);
        v16bf afrag;
        #pragma unroll
        for (int e = 0; e < 8; ++e) { afrag[e] = a_lo[e]; afrag[8 + e] = a_hi[e]; }
        #pragma unroll
        for (int t = 0; t < 2; ++t) {
            const float* wrow =
                Ww + (size_t)((nt0 + t) * 16 + l16) * DCAT + ks * 32 + half * 16;
            const f32x4* wp = (const f32x4*)wrow;
            f32x4 q0 = wp[0], q1 = wp[1], q2 = wp[2], q3 = wp[3];
            v16bf bfrag;
            #pragma unroll
            for (int e = 0; e < 4; ++e) {
                bfrag[e]      = (bf16)q0[e];
                bfrag[4 + e]  = (bf16)q1[e];
                bfrag[8 + e]  = (bf16)q2[e];
                bfrag[12 + e] = (bf16)q3[e];
            }
            acc2[t] = __builtin_amdgcn_wmma_f32_16x16x32_bf16(
                false, afrag, false, bfrag, (short)0, acc2[t], false, false);
        }
    }

    // ---- Stage 3: bias + LeakyReLU, row L2 norm, safediv, store ----
    const float bwv0 = bw[nt0 * 16 + l16];
    const float bwv1 = bw[(nt0 + 1) * 16 + l16];
    float h2v[2][8];
    float pj[8];
    #pragma unroll
    for (int j = 0; j < 8; ++j) {
        float v0 = leaky(acc2[0][j] + bwv0);
        float v1 = leaky(acc2[1][j] + bwv1);
        h2v[0][j] = v0; h2v[1][j] = v1;
        float p = v0 * v0 + v1 * v1;
        #pragma unroll
        for (int m = 1; m <= 8; m <<= 1) p += __shfl_xor(p, m, 32);  // reduce over l16
        pj[j] = p;
    }
    if (l16 == 0) {                       // lane 0 -> rows 0..7, lane 16 -> rows 8..15
        #pragma unroll
        for (int j = 0; j < 8; ++j)
            atomicAdd(&norm2[half * 8 + j], pj[j]);   // ds_add_f32 across 4 waves
    }
    __syncthreads();

    #pragma unroll
    for (int j = 0; j < 8; ++j) {
        const int row = half * 8 + j;
        const int g = blockBase + row;
        if (g < Nn) {
            float nrm = sqrtf(norm2[row]);
            float d = (nrm == 0.f) ? 1.f : nrm;
            out[(size_t)g * DOUT + nt0 * 16 + l16]       = h2v[0][j] / d;
            out[(size_t)g * DOUT + (nt0 + 1) * 16 + l16] = h2v[1][j] / d;
        }
    }
}

extern "C" void kernel_launch(void* const* d_in, const int* in_sizes, int n_in,
                              void* d_out, int out_size, void* d_ws, size_t ws_size,
                              hipStream_t stream) {
    const float* h_node  = (const float*)d_in[0];
    const float* h_ngbrs = (const float*)d_in[1];
    const float* alpha   = (const float*)d_in[2];
    const float* Wq      = (const float*)d_in[3];
    const float* bq      = (const float*)d_in[4];
    const float* Ww      = (const float*)d_in[5];
    const float* bw      = (const float*)d_in[6];
    float* out = (float*)d_out;

    const int Nn = in_sizes[0] / DIN;                       // 50000
    const int blocks = (Nn + NODES_PER_BLOCK - 1) / NODES_PER_BLOCK;  // 3125
    pinsage_fused_kernel<<<blocks, 128, 0, stream>>>(
        h_node, h_ngbrs, alpha, Wq, bq, Ww, bw, out, Nn);
}
